// LlamaAttention_83442624627158
// MI455X (gfx1250) — matrix operand order repaired and checked by reading, not compiled, not run on hardware
//
#include <hip/hip_runtime.h>
#include <hip/hip_bf16.h>

// ---------------------------------------------------------------------------
// Types for CDNA5 WMMA builtins (wave32)
// ---------------------------------------------------------------------------
typedef __attribute__((ext_vector_type(16))) __bf16 v16bf;
typedef __attribute__((ext_vector_type(8)))  __bf16 v8bf;
typedef __attribute__((ext_vector_type(4)))  __bf16 v4bf;
typedef __attribute__((ext_vector_type(8)))  float  v8f;

#define MTOT 4096   // B*S rows
#define NTOT 2048   // model dim
#define KTOT 2048   // model dim
#define SEQ  2048
#define HDIM 128

// ---------------------------------------------------------------------------
// bf16 split helpers (used only in one-time conversion / store paths)
// ---------------------------------------------------------------------------
__device__ __forceinline__ unsigned short bf16_bits(float f) {
    unsigned u = __builtin_bit_cast(unsigned, f);
    unsigned r = u + 0x7fffu + ((u >> 16) & 1u);
    return (unsigned short)(r >> 16);
}
__device__ __forceinline__ float bf16_to_f32(unsigned short b) {
    unsigned u = ((unsigned)b) << 16;
    return __builtin_bit_cast(float, u);
}
__device__ __forceinline__ __bf16 bf16_from_bits(unsigned short b) {
    return __builtin_bit_cast(__bf16, b);
}
__device__ __forceinline__ void split_bf16(float x, __bf16& hi, __bf16& lo) {
    unsigned short h = bf16_bits(x);
    float rem = x - bf16_to_f32(h);
    hi = bf16_from_bits(h);
    lo = bf16_from_bits(bf16_bits(rem));
}
__device__ __forceinline__ v8f wmma_bf16(v16bf a, v16bf b, v8f c) {
    return __builtin_amdgcn_wmma_f32_16x16x32_bf16(
        false, a, false, b, (short)0, c, false, false);
}
// A-matrix fragment from a bf16 plane: runs [p..p+7] and [p+16..p+23]
__device__ __forceinline__ v16bf load_afrag(const __bf16* p) {
    v8bf a = *(const v8bf*)p;
    v8bf b = *(const v8bf*)(p + 16);
    return __builtin_shufflevector(a, b, 0,1,2,3,4,5,6,7,8,9,10,11,12,13,14,15);
}
// B-matrix fragment: runs [p..p+7] and [p+16..p+23]
__device__ __forceinline__ v16bf load_bfrag(const __bf16* p) {
    v8bf a = *(const v8bf*)p;
    v8bf b = *(const v8bf*)(p + 16);
    return __builtin_shufflevector(a, b, 0,1,2,3,4,5,6,7,8,9,10,11,12,13,14,15);
}

// ---------------------------------------------------------------------------
// One-pass fp32 -> (hi, lo) bf16 plane conversion
// ---------------------------------------------------------------------------
__global__ __launch_bounds__(256) void split_f32(
    const float* __restrict__ in, __bf16* __restrict__ hi,
    __bf16* __restrict__ lo, int n4)
{
    const int i = blockIdx.x * 256 + threadIdx.x;
    if (i >= n4) return;
    float4 f = ((const float4*)in)[i];
    __bf16 h0, h1, h2, h3, l0, l1, l2, l3;
    split_bf16(f.x, h0, l0);
    split_bf16(f.y, h1, l1);
    split_bf16(f.z, h2, l2);
    split_bf16(f.w, h3, l3);
    v4bf h = {h0, h1, h2, h3};
    v4bf l = {l0, l1, l2, l3};
    ((v4bf*)hi)[i] = h;
    ((v4bf*)lo)[i] = l;
}

// ---------------------------------------------------------------------------
// GEMM: Y[M,N] = X[M,K] @ W[N,K]^T, all operands as pre-split bf16 planes.
// 8 waves/WG; wave tile = 32(M) x 64(N); WG tile = 64 x 256.
// mode 0: store fp32 row-major into Yf
// mode 1: store split planes row-major into Yh/Yl (scaled)
// mode 2: store split planes transposed [N, M] into Yh/Yl (scaled)
// ---------------------------------------------------------------------------
__global__ __launch_bounds__(256) void gemm_planes(
    const __bf16* __restrict__ Xh, const __bf16* __restrict__ Xl,
    const __bf16* __restrict__ Wh, const __bf16* __restrict__ Wl,
    float* __restrict__ Yf,
    __bf16* __restrict__ Yh, __bf16* __restrict__ Yl,
    int mode, float scale)
{
    const int lane = threadIdx.x & 31;
    const int wave = threadIdx.x >> 5;
    const int wm   = wave & 1;
    const int wn   = wave >> 1;
    const int half = lane >> 4;
    const int l15  = lane & 15;
    const int m0   = blockIdx.y * 64  + wm * 32;
    const int n0   = blockIdx.x * 256 + wn * 64;

    v8f acc[2][4];
#pragma unroll
    for (int i = 0; i < 2; ++i)
#pragma unroll
        for (int j = 0; j < 4; ++j)
            acc[i][j] = (v8f){0.f,0.f,0.f,0.f,0.f,0.f,0.f,0.f};

    for (int k0 = 0; k0 < KTOT; k0 += 32) {
        // ---- A fragments: direct bf16 plane loads, zero conversion VALU ----
        v16bf ah[2], al[2];
#pragma unroll
        for (int i = 0; i < 2; ++i) {
            const size_t off = (size_t)(m0 + i * 16 + l15) * KTOT + k0 + half * 8;
            ah[i] = load_afrag(Xh + off);
            al[i] = load_afrag(Xl + off);
            // unconditional speculative prefetch, no guard branches
            __builtin_prefetch(Xh + off + 32, 0, 0);
        }
        // ---- B fragments ----
        v16bf bh[4], bl[4];
#pragma unroll
        for (int j = 0; j < 4; ++j) {
            const size_t off = (size_t)(n0 + j * 16 + l15) * KTOT + k0 + half * 8;
            bh[j] = load_bfrag(Wh + off);
            bl[j] = load_bfrag(Wl + off);
            __builtin_prefetch(Wh + off + 32, 0, 0);
        }
        // ---- 3-product split accumulation: hi*hi + hi*lo + lo*hi ----
#pragma unroll
        for (int i = 0; i < 2; ++i)
#pragma unroll
            for (int j = 0; j < 4; ++j) {
                acc[i][j] = wmma_bf16(ah[i], bh[j], acc[i][j]);
                acc[i][j] = wmma_bf16(ah[i], bl[j], acc[i][j]);
                acc[i][j] = wmma_bf16(al[i], bh[j], acc[i][j]);
            }
    }

    // ---- store (C layout: elem r -> row r+8*half, col = lane&15) ----
#pragma unroll
    for (int i = 0; i < 2; ++i)
#pragma unroll
        for (int j = 0; j < 4; ++j)
#pragma unroll
            for (int r = 0; r < 8; ++r) {
                const int row = m0 + i * 16 + r + 8 * half;
                const int col = n0 + j * 16 + l15;
                const float v = acc[i][j][r] * scale;
                if (mode == 0) {
                    Yf[(size_t)row * NTOT + col] = v;
                } else {
                    __bf16 h, l;
                    split_bf16(v, h, l);
                    const size_t idx = (mode == 1)
                        ? (size_t)row * NTOT + col
                        : (size_t)col * MTOT + row;
                    Yh[idx] = h;
                    Yl[idx] = l;
                }
            }
}

// ---------------------------------------------------------------------------
// Flash attention (causal), one wave per 16-row q-tile per (b,h).
// Q,K planes row-major [4096,2048]; V planes transposed [2048,4096].
// Output written as split planes (feeds final projection GEMM).
// Scale 1/sqrt(128) already folded into Q planes by the Q-projection store.
// ---------------------------------------------------------------------------
__global__ __launch_bounds__(128) void flash_attn(
    const __bf16* __restrict__ Qh, const __bf16* __restrict__ Ql,
    const __bf16* __restrict__ Kh, const __bf16* __restrict__ Kl,
    const __bf16* __restrict__ Vth, const __bf16* __restrict__ Vtl,
    __bf16* __restrict__ Oh, __bf16* __restrict__ Ol)
{
    __shared__ unsigned pbuf[4][16][32];   // per-wave P staging (hi<<16 | lo)

    const int lane = threadIdx.x & 31;
    const int wave = threadIdx.x >> 5;
    const int half = lane >> 4;
    const int l15  = lane & 15;
    const int q0   = blockIdx.x * 64 + wave * 16;
    const int bh   = blockIdx.y;
    const int b    = bh >> 4;
    const int h    = bh & 15;
    const int rowbase = b * SEQ;
    const int cb      = h * HDIM;

    // ---- Q fragments (A layout) straight from planes ----
    v16bf qfh[4], qfl[4];
#pragma unroll
    for (int c = 0; c < 4; ++c) {
        const size_t off = (size_t)(rowbase + q0 + l15) * NTOT + cb + c * 32 + half * 8;
        qfh[c] = load_afrag(Qh + off);
        qfl[c] = load_afrag(Ql + off);
    }

    float m_r[8], l_r[8];
    v8f o[8];
#pragma unroll
    for (int r = 0; r < 8; ++r) { m_r[r] = -__builtin_inff(); l_r[r] = 0.f; }
#pragma unroll
    for (int dt = 0; dt < 8; ++dt) o[dt] = (v8f){0.f,0.f,0.f,0.f,0.f,0.f,0.f,0.f};

    const int kbmax = (q0 + 15) >> 5;
    for (int kb = 0; kb <= kbmax; ++kb) {
        const int key0 = kb * 32;
        v8f s0 = (v8f){0.f,0.f,0.f,0.f,0.f,0.f,0.f,0.f};
        v8f s1 = s0;

        // ---- scores: S = Q @ K^T, two 16-key tiles ----
#pragma unroll
        for (int c = 0; c < 4; ++c) {
            const size_t o0 = (size_t)(rowbase + key0 + l15)      * NTOT + cb + c * 32 + half * 8;
            const size_t o1 = (size_t)(rowbase + key0 + 16 + l15) * NTOT + cb + c * 32 + half * 8;
            v16bf k0h = load_bfrag(Kh + o0);
            v16bf k0l = load_bfrag(Kl + o0);
            v16bf k1h = load_bfrag(Kh + o1);
            v16bf k1l = load_bfrag(Kl + o1);
            s0 = wmma_bf16(qfh[c], k0h, s0);
            s0 = wmma_bf16(qfh[c], k0l, s0);
            s0 = wmma_bf16(qfl[c], k0h, s0);
            s1 = wmma_bf16(qfh[c], k1h, s1);
            s1 = wmma_bf16(qfh[c], k1l, s1);
            s1 = wmma_bf16(qfl[c], k1h, s1);
        }

        // ---- causal mask (diagonal block only) ----
        if (key0 + 31 > q0) {
#pragma unroll
            for (int r = 0; r < 8; ++r) {
                const int row = q0 + r + 8 * half;
                if (key0 + l15      > row) s0[r] = -1e30f;
                if (key0 + 16 + l15 > row) s1[r] = -1e30f;
            }
        }

        // ---- online softmax (rows live across 16-lane groups) ----
        float alpha[8];
#pragma unroll
        for (int r = 0; r < 8; ++r) {
            float t = fmaxf(s0[r], s1[r]);
            t = fmaxf(t, __shfl_xor(t, 1, 32));
            t = fmaxf(t, __shfl_xor(t, 2, 32));
            t = fmaxf(t, __shfl_xor(t, 4, 32));
            t = fmaxf(t, __shfl_xor(t, 8, 32));
            const float mnew = fmaxf(m_r[r], t);
            const float a  = __expf(m_r[r] - mnew);
            const float p0 = __expf(s0[r] - mnew);
            const float p1 = __expf(s1[r] - mnew);
            float rs = p0 + p1;
            rs += __shfl_xor(rs, 1, 32);
            rs += __shfl_xor(rs, 2, 32);
            rs += __shfl_xor(rs, 4, 32);
            rs += __shfl_xor(rs, 8, 32);
            l_r[r] = l_r[r] * a + rs;
            m_r[r] = mnew;
            alpha[r] = a;
            // stage P (bf16 split, packed) through LDS: C layout -> A layout
            const unsigned short h0 = bf16_bits(p0);
            const unsigned short g0 = bf16_bits(p0 - bf16_to_f32(h0));
            pbuf[wave][r + 8 * half][l15]      = ((unsigned)h0 << 16) | (unsigned)g0;
            const unsigned short h1 = bf16_bits(p1);
            const unsigned short g1 = bf16_bits(p1 - bf16_to_f32(h1));
            pbuf[wave][r + 8 * half][16 + l15] = ((unsigned)h1 << 16) | (unsigned)g1;
        }

        // ---- rescale running output ----
#pragma unroll
        for (int dt = 0; dt < 8; ++dt)
#pragma unroll
            for (int r = 0; r < 8; ++r) o[dt][r] *= alpha[r];

        __asm__ volatile("" ::: "memory");  // keep LDS write->read ordered (in-order per wave)

        // ---- reload P as A fragment (16q x 32k) ----
        v16bf ph, pl;
#pragma unroll
        for (int e = 0; e < 8; ++e) {
            const unsigned w0 = pbuf[wave][l15][half * 8 + e];
            ph[e]     = bf16_from_bits((unsigned short)(w0 >> 16));
            pl[e]     = bf16_from_bits((unsigned short)(w0 & 0xffffu));
            const unsigned w1 = pbuf[wave][l15][16 + half * 8 + e];
            ph[8 + e] = bf16_from_bits((unsigned short)(w1 >> 16));
            pl[8 + e] = bf16_from_bits((unsigned short)(w1 & 0xffffu));
        }

        // ---- O += P @ V  (V^T plane rows contiguous in key) ----
#pragma unroll
        for (int dt = 0; dt < 8; ++dt) {
            const size_t off = (size_t)(cb + dt * 16 + l15) * MTOT
                             + rowbase + key0 + half * 8;
            v16bf vh = load_bfrag(Vth + off);
            v16bf vl = load_bfrag(Vtl + off);
            o[dt] = wmma_bf16(ph, vh, o[dt]);
            o[dt] = wmma_bf16(ph, vl, o[dt]);
            o[dt] = wmma_bf16(pl, vh, o[dt]);
        }
    }

    // ---- finalize: divide by denom, store as split planes ----
    float invl[8];
#pragma unroll
    for (int r = 0; r < 8; ++r) invl[r] = 1.0f / l_r[r];
#pragma unroll
    for (int dt = 0; dt < 8; ++dt)
#pragma unroll
        for (int r = 0; r < 8; ++r) {
            const int row = rowbase + q0 + r + 8 * half;
            const int col = cb + dt * 16 + l15;
            __bf16 hh, ll;
            split_bf16(o[dt][r] * invl[r], hh, ll);
            Oh[(size_t)row * NTOT + col] = hh;
            Ol[(size_t)row * NTOT + col] = ll;
        }
}

// ---------------------------------------------------------------------------
// Launch: split x once; per weight: split -> GEMM; attention; final GEMM.
// inputs: [0]=x, [1]=mask(unused; causal inline), [2..5]=Wq,Wk,Wv,Wo
// Workspace (144 MB): Xh,Xl (reused as attention-output planes), shared
// Wh,Wl, Qh,Ql, Kh,Kl, Vth,Vtl -- all bf16, L2-resident on 192 MB L2.
// ---------------------------------------------------------------------------
extern "C" void kernel_launch(void* const* d_in, const int* in_sizes, int n_in,
                              void* d_out, int out_size, void* d_ws, size_t ws_size,
                              hipStream_t stream) {
    (void)in_sizes; (void)n_in; (void)out_size; (void)ws_size;
    const float* x  = (const float*)d_in[0];
    const float* Wq = (const float*)d_in[2];
    const float* Wk = (const float*)d_in[3];
    const float* Wv = (const float*)d_in[4];
    const float* Wo = (const float*)d_in[5];

    const size_t nMK = (size_t)MTOT * KTOT;   // 8M elements
    const size_t nNK = (size_t)NTOT * KTOT;   // 4M elements

    char* p = (char*)d_ws;
    __bf16* Xh  = (__bf16*)p;               p += nMK * 2;
    __bf16* Xl  = (__bf16*)p;               p += nMK * 2;
    __bf16* Wh  = (__bf16*)p;               p += nNK * 2;
    __bf16* Wl  = (__bf16*)p;               p += nNK * 2;
    __bf16* Qh  = (__bf16*)p;               p += nMK * 2;
    __bf16* Ql  = (__bf16*)p;               p += nMK * 2;
    __bf16* Kh  = (__bf16*)p;               p += nMK * 2;
    __bf16* Kl  = (__bf16*)p;               p += nMK * 2;
    __bf16* Vth = (__bf16*)p;               p += nMK * 2;
    __bf16* Vtl = (__bf16*)p;               p += nMK * 2;
    // attention-output planes alias X planes (X fully consumed by then)
    __bf16* Ah = Xh;
    __bf16* Al = Xl;

    const int xb = (int)(nMK / 4 / 256);      // 8192 blocks
    const int wb = (int)(nNK / 4 / 256);      // 4096 blocks
    dim3 ggrid(NTOT / 256, MTOT / 64);        // (8, 64)
    const float qscale = 0.08838834764831845f;  // 1/sqrt(128)

    split_f32<<<xb, 256, 0, stream>>>(x, Xh, Xl, (int)(nMK / 4));

    split_f32<<<wb, 256, 0, stream>>>(Wq, Wh, Wl, (int)(nNK / 4));
    gemm_planes<<<ggrid, 256, 0, stream>>>(Xh, Xl, Wh, Wl, nullptr, Qh, Ql, 1, qscale);

    split_f32<<<wb, 256, 0, stream>>>(Wk, Wh, Wl, (int)(nNK / 4));
    gemm_planes<<<ggrid, 256, 0, stream>>>(Xh, Xl, Wh, Wl, nullptr, Kh, Kl, 1, 1.0f);

    split_f32<<<wb, 256, 0, stream>>>(Wv, Wh, Wl, (int)(nNK / 4));
    gemm_planes<<<ggrid, 256, 0, stream>>>(Xh, Xl, Wh, Wl, nullptr, Vth, Vtl, 2, 1.0f);

    flash_attn<<<dim3(SEQ / 64, 32), 128, 0, stream>>>(Qh, Ql, Kh, Kl, Vth, Vtl, Ah, Al);

    split_f32<<<wb, 256, 0, stream>>>(Wo, Wh, Wl, (int)(nNK / 4));
    gemm_planes<<<ggrid, 256, 0, stream>>>(Ah, Al, Wh, Wl, (float*)d_out, nullptr, nullptr, 0, 1.0f);
}
